// SATokenProjector_3109556322891
// MI455X (gfx1250) — matrix operand order repaired and checked by reading, not compiled, not run on hardware
//
#include <hip/hip_runtime.h>

typedef __bf16 bf16;
typedef float  v8f  __attribute__((ext_vector_type(8)));
typedef bf16   v8bf __attribute__((ext_vector_type(8)));
typedef bf16   v16bf __attribute__((ext_vector_type(16)));

#define B_   32
#define N1_  2048
#define DIM_ 1024
#define HEADS_ 16
#define DH_  64
#define NQ_  64
#define L_   2
#define FF_  4096
#define NKV_ 2112           // N1 + NQ
#define KV2_ 2048           // 2 * INNER

__device__ inline v16bf pack16(v8bf lo, v8bf hi) {
  v16bf r;
#pragma unroll
  for (int i = 0; i < 8; ++i) { r[i] = lo[i]; r[8 + i] = hi[i]; }
  return r;
}

__device__ inline v8f wmma_bf16(v16bf a, v16bf b, v8f c) {
  return __builtin_amdgcn_wmma_f32_16x16x32_bf16(false, a, false, b, (short)0, c,
                                                 false, false);
}

__device__ inline float gelu_tanh(float x) {
  float x3 = x * x * x;
  return 0.5f * x * (1.f + tanhf(0.7978845608028654f * (x + 0.044715f * x3)));
}

// Async copy of 32 contiguous bytes global -> LDS (2 x b128), ASYNCcnt-tracked.
__device__ inline void async_copy32(unsigned lds_addr, const bf16* gptr) {
  asm volatile(
      "global_load_async_to_lds_b128 %0, %1, off\n\t"
      "global_load_async_to_lds_b128 %0, %1, off offset:16"
      :: "v"(lds_addr), "v"((unsigned long long)(uintptr_t)gptr)
      : "memory");
}

__device__ inline void wait_async0() {
  asm volatile("s_wait_asynccnt 0" ::: "memory");
}

// ---------------------------------------------------------------- elementwise
__global__ __launch_bounds__(256) void cast_bf16_kernel(const float* __restrict__ in,
                                                        bf16* __restrict__ out, int n) {
  int i = blockIdx.x * 256 + threadIdx.x;
  if (i < n) out[i] = (bf16)in[i];
}

__global__ __launch_bounds__(256) void bcast_lat_kernel(const float* __restrict__ latents,
                                                        float* __restrict__ lat) {
  int i = blockIdx.x * 256 + threadIdx.x;
  if (i < B_ * NQ_ * DIM_) lat[i] = latents[i & (NQ_ * DIM_ - 1)];
}

// ---------------------------------------------------------------- layernorm
// One block per input row (1024 cols). Output row remap:
//   orow = (row / rpb) * obatch + ooff + row % rpb
template <typename OutT>
__global__ __launch_bounds__(256) void ln_kernel(const float* __restrict__ in,
                                                 const float* __restrict__ sc,
                                                 const float* __restrict__ bi,
                                                 OutT* __restrict__ out,
                                                 int rpb, int obatch, int ooff) {
  __shared__ float red[256];
  __shared__ float red2[256];
  int row = blockIdx.x, tid = threadIdx.x;
  const float4 v = *(const float4*)(in + (size_t)row * DIM_ + tid * 4);
  float s  = v.x + v.y + v.z + v.w;
  float s2 = v.x * v.x + v.y * v.y + v.z * v.z + v.w * v.w;
  red[tid] = s; red2[tid] = s2;
  __syncthreads();
#pragma unroll
  for (int off = 128; off > 0; off >>= 1) {
    if (tid < off) { red[tid] += red[tid + off]; red2[tid] += red2[tid + off]; }
    __syncthreads();
  }
  float mean = red[0] * (1.f / DIM_);
  float var  = red2[0] * (1.f / DIM_) - mean * mean;
  float inv  = rsqrtf(var + 1e-5f);
  int orow = (row / rpb) * obatch + ooff + row % rpb;
  const float4 g  = *(const float4*)(sc + tid * 4);
  const float4 bb = *(const float4*)(bi + tid * 4);
  OutT* q = out + (size_t)orow * DIM_ + tid * 4;
  q[0] = (OutT)((v.x - mean) * inv * g.x + bb.x);
  q[1] = (OutT)((v.y - mean) * inv * g.y + bb.y);
  q[2] = (OutT)((v.z - mean) * inv * g.z + bb.z);
  q[3] = (OutT)((v.w - mean) * inv * g.w + bb.w);
}

// ---------------------------------------------------------------- bf16 GEMM
// C[M,N] = A[M,K] @ B[K,N], bf16 in, fp32 accumulate via v_wmma_f32_16x16x32_bf16.
// Block tile 128x64, BK=32, 256 threads = 8 waves in a 4x2 grid; each wave owns a
// 32x32 output tile (2 A-frags x 2 B-frags -> 4 WMMAs per k-step, each fragment
// reused twice). A tile is copied global->LDS with GLOBAL_LOAD_ASYNC_TO_LDS_B128
// (ASYNCcnt); B tile is transposed through VGPRs into LDS.
// EPI: 0 = store bf16 | 1 = gelu->bf16 | 2 = aux_f32[idx] += acc | 3 = f32 + bias
template <int EPI>
__global__ __launch_bounds__(256) void gemm_kernel(const bf16* __restrict__ A,
                                                   const bf16* __restrict__ Bm,
                                                   void* __restrict__ Cout,
                                                   const float* __restrict__ bias,
                                                   int M, int N, int K,
                                                   int a_rpb, int a_batch, int a_off) {
  __shared__ __align__(16) bf16 As[128][40];  // [m][k]
  __shared__ __align__(16) bf16 Bs[64][40];   // [n][k] (transposed)
  int tid = threadIdx.x, lane = tid & 31, wave = tid >> 5;
  int wm = wave >> 1, wn = wave & 1;          // 4 x 2 wave grid
  int m0 = blockIdx.y * 128, n0 = blockIdx.x * 64;
  int fm = lane & 15, kh = lane >> 4;

  v8f acc[2][2];
#pragma unroll
  for (int mi = 0; mi < 2; ++mi)
#pragma unroll
    for (int ni = 0; ni < 2; ++ni)
#pragma unroll
      for (int i = 0; i < 8; ++i) acc[mi][ni][i] = 0.f;

  // A tile: 128 rows x 32 halves; thread t copies 32B (half a row).
  int ar = tid >> 1, acol = (tid & 1) * 16;
  int grow = m0 + ar;
  size_t arow = (size_t)((grow / a_rpb) * a_batch + a_off + grow % a_rpb);
  const bf16* agp = A + arow * (size_t)K + acol;
  unsigned alds = (unsigned)(uintptr_t)&As[ar][acol];
  // B tile: 32 k x 64 n; thread t loads 8 n's at one k, stores transposed.
  int bk = tid >> 3, bn = (tid & 7) * 8;

  for (int k0 = 0; k0 < K; k0 += 32) {
    __syncthreads();
    async_copy32(alds, agp + k0);
    v8bf bv = *(const v8bf*)(Bm + (size_t)(k0 + bk) * N + n0 + bn);
#pragma unroll
    for (int i = 0; i < 8; ++i) Bs[bn + i][bk] = bv[i];
    wait_async0();
    __syncthreads();

    v16bf af[2], bfr[2];
#pragma unroll
    for (int mi = 0; mi < 2; ++mi) {
      int mrow = wm * 32 + mi * 16 + fm;
      v8bf lo = *(const v8bf*)&As[mrow][kh * 8];
      v8bf hi = *(const v8bf*)&As[mrow][16 + kh * 8];
      af[mi] = pack16(lo, hi);
    }
#pragma unroll
    for (int ni = 0; ni < 2; ++ni) {
      int nrow = wn * 32 + ni * 16 + fm;
      v8bf lo = *(const v8bf*)&Bs[nrow][kh * 8];
      v8bf hi = *(const v8bf*)&Bs[nrow][16 + kh * 8];
      bfr[ni] = pack16(lo, hi);
    }
#pragma unroll
    for (int mi = 0; mi < 2; ++mi)
#pragma unroll
      for (int ni = 0; ni < 2; ++ni)
        acc[mi][ni] = wmma_bf16(af[mi], bfr[ni], acc[mi][ni]);
  }

#pragma unroll
  for (int mi = 0; mi < 2; ++mi) {
    int rbase = m0 + wm * 32 + mi * 16 + 8 * kh;
#pragma unroll
    for (int ni = 0; ni < 2; ++ni) {
      int col = n0 + wn * 32 + ni * 16 + fm;
#pragma unroll
      for (int v = 0; v < 8; ++v) {
        size_t idx = (size_t)(rbase + v) * N + col;
        float val = acc[mi][ni][v];
        if (EPI == 0) {
          ((bf16*)Cout)[idx] = (bf16)val;
        } else if (EPI == 1) {
          ((bf16*)Cout)[idx] = (bf16)gelu_tanh(val);
        } else if (EPI == 2) {
          ((float*)Cout)[idx] += val;
        } else {
          ((float*)Cout)[idx] = val + bias[col];
        }
      }
    }
  }
}

// ---------------------------------------------------------------- attention
// 1 block per (batch, head); 128 threads = 4 waves, wave w handles query rows
// [w*16, w*16+16). Streaming softmax over key chunks of 32.
__global__ __launch_bounds__(128) void attn_kernel(const bf16* __restrict__ qb,
                                                   const bf16* __restrict__ kvb,
                                                   bf16* __restrict__ ob) {
  __shared__ __align__(16) bf16 vT[64][40];          // [dh][key] transposed V chunk
  __shared__ __align__(16) bf16 Pb[4][16][32];       // per-wave P repack buffer
  int b = blockIdx.x >> 4, h = blockIdx.x & 15;
  int tid = threadIdx.x, wave = tid >> 5, lane = tid & 31;
  int fm = lane & 15, kh = lane >> 4;
  int nq0 = wave * 16;

  // Q fragments (16x64 -> two 16x32 A-frags), loaded once.
  const bf16* qbase = qb + (size_t)(b * NQ_ + nq0 + fm) * DIM_ + h * DH_;
  v16bf aq[2];
#pragma unroll
  for (int j = 0; j < 2; ++j) {
    v8bf lo = *(const v8bf*)(qbase + j * 32 + kh * 8);
    v8bf hi = *(const v8bf*)(qbase + j * 32 + 16 + kh * 8);
    aq[j] = pack16(lo, hi);
  }

  float mrow[8], lrow[8];
  v8f oacc[4];
#pragma unroll
  for (int v = 0; v < 8; ++v) { mrow[v] = -3.0e38f; lrow[v] = 0.f; }
#pragma unroll
  for (int nt = 0; nt < 4; ++nt)
#pragma unroll
    for (int v = 0; v < 8; ++v) oacc[nt][v] = 0.f;

  const bf16* kvbase = kvb + (size_t)b * NKV_ * KV2_ + h * DH_;

  for (int c0 = 0; c0 < NKV_; c0 += 32) {
    __syncthreads();
    {  // cooperative V chunk load, transposed into LDS
      int key = tid >> 2, dh0 = (tid & 3) * 16;
      const bf16* vp = kvbase + (size_t)(c0 + key) * KV2_ + 1024 + dh0;
      v8bf x0 = *(const v8bf*)(vp);
      v8bf x1 = *(const v8bf*)(vp + 8);
#pragma unroll
      for (int i = 0; i < 8; ++i) { vT[dh0 + i][key] = x0[i]; vT[dh0 + 8 + i][key] = x1[i]; }
    }
    __syncthreads();

    // S = Q K^T (two 16-key tiles, K-dim 64 = 2 chained WMMAs each)
    v8f s0, s1;
#pragma unroll
    for (int i = 0; i < 8; ++i) { s0[i] = 0.f; s1[i] = 0.f; }
#pragma unroll
    for (int t = 0; t < 2; ++t) {
      const bf16* kp = kvbase + (size_t)(c0 + t * 16 + fm) * KV2_;
#pragma unroll
      for (int j = 0; j < 2; ++j) {
        v8bf lo = *(const v8bf*)(kp + j * 32 + kh * 8);
        v8bf hi = *(const v8bf*)(kp + j * 32 + 16 + kh * 8);
        v16bf bk = pack16(lo, hi);
        if (t == 0) s0 = wmma_bf16(aq[j], bk, s0);
        else        s1 = wmma_bf16(aq[j], bk, s1);
      }
    }

    // online softmax update, rows spread over 16-lane half-groups
#pragma unroll
    for (int v = 0; v < 8; ++v) {
      float a0 = s0[v] * 0.125f, a1 = s1[v] * 0.125f;   // (DH^-0.25)^2 = 1/8
      float rm = fmaxf(a0, a1);
#pragma unroll
      for (int off = 1; off < 16; off <<= 1) rm = fmaxf(rm, __shfl_xor(rm, off, 32));
      float mnew = fmaxf(mrow[v], rm);
      float corr = __expf(mrow[v] - mnew);
      float p0 = __expf(a0 - mnew), p1 = __expf(a1 - mnew);
      float rs = p0 + p1;
#pragma unroll
      for (int off = 1; off < 16; off <<= 1) rs += __shfl_xor(rs, off, 32);
      lrow[v] = lrow[v] * corr + rs;
      mrow[v] = mnew;
      int r = v + 8 * kh;
      Pb[wave][r][fm]      = (bf16)p0;
      Pb[wave][r][16 + fm] = (bf16)p1;
#pragma unroll
      for (int nt = 0; nt < 4; ++nt) oacc[nt][v] *= corr;
    }
    asm volatile("s_wait_dscnt 0" ::: "memory");   // wave-local LDS RAW

    // repack P (16x32) as A-frag, accumulate O += P V
    v8bf plo = *(const v8bf*)&Pb[wave][fm][kh * 8];
    v8bf phi = *(const v8bf*)&Pb[wave][fm][16 + kh * 8];
    v16bf ap = pack16(plo, phi);
#pragma unroll
    for (int nt = 0; nt < 4; ++nt) {
      v8bf bl = *(const v8bf*)&vT[nt * 16 + fm][kh * 8];
      v8bf bh = *(const v8bf*)&vT[nt * 16 + fm][16 + kh * 8];
      oacc[nt] = wmma_bf16(ap, pack16(bl, bh), oacc[nt]);
    }
  }

  // normalize + store (row = b*64 + nq0 + v + 8*kh, col = h*64 + nt*16 + fm)
#pragma unroll
  for (int nt = 0; nt < 4; ++nt)
#pragma unroll
    for (int v = 0; v < 8; ++v) {
      int row = b * NQ_ + nq0 + v + 8 * kh;
      int col = h * DH_ + nt * 16 + fm;
      ob[(size_t)row * DIM_ + col] = (bf16)(oacc[nt][v] / lrow[v]);
    }
}

// ---------------------------------------------------------------- host
extern "C" void kernel_launch(void* const* d_in, const int* in_sizes, int n_in,
                              void* d_out, int out_size, void* d_ws, size_t ws_size,
                              hipStream_t stream) {
  (void)in_sizes; (void)n_in; (void)out_size; (void)ws_size;
  const float* x        = (const float*)d_in[0];
  const float* latents  = (const float*)d_in[1];
  const float* n1s = (const float*)d_in[2];
  const float* n1b = (const float*)d_in[3];
  const float* n2s = (const float*)d_in[4];
  const float* n2b = (const float*)d_in[5];
  const float* Wq  = (const float*)d_in[6];
  const float* Wkv = (const float*)d_in[7];
  const float* Wo  = (const float*)d_in[8];
  const float* ffs = (const float*)d_in[9];
  const float* ffb = (const float*)d_in[10];
  const float* W1  = (const float*)d_in[11];
  const float* W2  = (const float*)d_in[12];
  const float* Pw  = (const float*)d_in[13];
  const float* Pb_ = (const float*)d_in[14];
  const float* os  = (const float*)d_in[15];
  const float* obv = (const float*)d_in[16];
  float* out = (float*)d_out;

  size_t cur = 0;
  char* base = (char*)d_ws;
  auto carve = [&](size_t bytes) -> void* {
    cur = (cur + 255) & ~(size_t)255;
    void* p = base + cur;
    cur += bytes;
    return p;
  };
  const size_t E = sizeof(bf16);
  bf16* wq_bf  = (bf16*)carve((size_t)L_ * DIM_ * DIM_ * E);
  bf16* wkv_bf = (bf16*)carve((size_t)L_ * DIM_ * KV2_ * E);
  bf16* wo_bf  = (bf16*)carve((size_t)L_ * DIM_ * DIM_ * E);
  bf16* w1_bf  = (bf16*)carve((size_t)L_ * DIM_ * FF_ * E);
  bf16* w2_bf  = (bf16*)carve((size_t)L_ * FF_ * DIM_ * E);
  bf16* wp_bf  = (bf16*)carve((size_t)DIM_ * DIM_ * E);
  bf16* cat    = (bf16*)carve((size_t)B_ * NKV_ * DIM_ * E);
  bf16* kvbuf  = (bf16*)carve((size_t)B_ * NKV_ * KV2_ * E);
  bf16* qbuf   = (bf16*)carve((size_t)B_ * NQ_ * DIM_ * E);
  bf16* attnb  = (bf16*)carve((size_t)B_ * NQ_ * DIM_ * E);
  bf16* ffin   = (bf16*)carve((size_t)B_ * NQ_ * DIM_ * E);
  bf16* hbuf   = (bf16*)carve((size_t)B_ * NQ_ * FF_ * E);
  bf16* latbf  = (bf16*)carve((size_t)B_ * NQ_ * DIM_ * E);
  float* latf  = (float*)carve((size_t)B_ * NQ_ * DIM_ * 4);
  float* projf = (float*)carve((size_t)B_ * NQ_ * DIM_ * 4);

  auto cast = [&](const float* src, bf16* dst, int n) {
    cast_bf16_kernel<<<(n + 255) / 256, 256, 0, stream>>>(src, dst, n);
  };
  cast(Wq,  wq_bf,  L_ * DIM_ * DIM_);
  cast(Wkv, wkv_bf, L_ * DIM_ * KV2_);
  cast(Wo,  wo_bf,  L_ * DIM_ * DIM_);
  cast(W1,  w1_bf,  L_ * DIM_ * FF_);
  cast(W2,  w2_bf,  L_ * FF_ * DIM_);
  cast(Pw,  wp_bf,  DIM_ * DIM_);
  bcast_lat_kernel<<<(B_ * NQ_ * DIM_ + 255) / 256, 256, 0, stream>>>(latents, latf);

  const int Mlat = B_ * NQ_;           // 2048 latent rows
  const int Mkv  = B_ * NKV_;          // 67584 kv rows
  for (int i = 0; i < L_; ++i) {
    // LN(x) -> cat rows [0,2048) per batch ; LN(lat) -> cat rows [2048,2112)
    ln_kernel<bf16><<<B_ * N1_, 256, 0, stream>>>(x, n1s + i * DIM_, n1b + i * DIM_,
                                                  cat, N1_, NKV_, 0);
    ln_kernel<bf16><<<Mlat, 256, 0, stream>>>(latf, n2s + i * DIM_, n2b + i * DIM_,
                                              cat, NQ_, NKV_, N1_);
    // q = ltn @ Wq   (A = ltn rows inside cat, remapped)
    gemm_kernel<0><<<dim3(DIM_ / 64, Mlat / 128), 256, 0, stream>>>(
        cat, wq_bf + (size_t)i * DIM_ * DIM_, qbuf, nullptr,
        Mlat, DIM_, DIM_, NQ_, NKV_, N1_);
    // kv = cat @ Wkv  (the big GEMM: 67584 x 2048 x 1024)
    gemm_kernel<0><<<dim3(KV2_ / 64, Mkv / 128), 256, 0, stream>>>(
        cat, wkv_bf + (size_t)i * DIM_ * KV2_, kvbuf, nullptr,
        Mkv, KV2_, DIM_, Mkv, 0, 0);
    // attention
    attn_kernel<<<B_ * HEADS_, 128, 0, stream>>>(qbuf, kvbuf, attnb);
    // lat += attn @ Wo
    gemm_kernel<2><<<dim3(DIM_ / 64, Mlat / 128), 256, 0, stream>>>(
        attnb, wo_bf + (size_t)i * DIM_ * DIM_, latf, nullptr,
        Mlat, DIM_, DIM_, Mlat, 0, 0);
    // FFN
    ln_kernel<bf16><<<Mlat, 256, 0, stream>>>(latf, ffs + i * DIM_, ffb + i * DIM_,
                                              ffin, Mlat, 0, 0);
    gemm_kernel<1><<<dim3(FF_ / 64, Mlat / 128), 256, 0, stream>>>(
        ffin, w1_bf + (size_t)i * DIM_ * FF_, hbuf, nullptr,
        Mlat, FF_, DIM_, Mlat, 0, 0);
    gemm_kernel<2><<<dim3(DIM_ / 64, Mlat / 128), 256, 0, stream>>>(
        hbuf, w2_bf + (size_t)i * FF_ * DIM_, latf, nullptr,
        Mlat, DIM_, FF_, Mlat, 0, 0);
  }

  // head: out = LN(lat @ proj_w + proj_b)
  cast_bf16_kernel<<<(Mlat * DIM_ + 255) / 256, 256, 0, stream>>>(latf, latbf, Mlat * DIM_);
  gemm_kernel<3><<<dim3(DIM_ / 64, Mlat / 128), 256, 0, stream>>>(
      latbf, wp_bf, projf, Pb_, Mlat, DIM_, DIM_, Mlat, 0, 0);
  ln_kernel<float><<<Mlat, 256, 0, stream>>>(projf, os, obv, out, Mlat, 0, 0);
}
